// YATE_Attention_41875931136320
// MI455X (gfx1250) — matrix operand order repaired and checked by reading, not compiled, not run on hardware
//
#include <hip/hip_runtime.h>

typedef __bf16 bf16_t;
typedef __bf16 v16bf __attribute__((ext_vector_type(16)));
typedef float  v8f   __attribute__((ext_vector_type(8)));

static constexpr int NN = 100000;
static constexpr int EE = 1600000;
static constexpr int DD = 64;

// ---- workspace layout (bytes) ----
static constexpr size_t OFF_WT     = 0;                       // 4 x 64x64 bf16 (transposed) = 32768 B
static constexpr size_t OFF_Q      = 32768;                   // N*64 f32 = 25,600,000 B
static constexpr size_t OFF_SCORES = 25632768;                // E*4  f32 = 25,600,000 B
static constexpr size_t OFF_MKEYS  = 51232768;                // N*4  u32 =  1,600,000 B
static constexpr size_t OFF_DENOM  = 52832768;                // N*4  f32 =  1,600,000 B
static constexpr size_t OFF_ACCUM  = 54432768;                // N*64 f32 = 25,600,000 B  (end ~80MB)

// monotonic uint encoding of float for atomicMax-based segment max
__device__ __forceinline__ unsigned fkey(float f) {
  unsigned b = __float_as_uint(f);
  return (b & 0x80000000u) ? ~b : (b | 0x80000000u);
}
__device__ __forceinline__ float fdecode(unsigned k) {
  return (k & 0x80000000u) ? __uint_as_float(k & 0x7FFFFFFFu) : __uint_as_float(~k);
}

// A fragment (16x32 bf16) for Z = ea * xn, row m = lane%16, callers give k = 16*half + s (contiguous)
__device__ __forceinline__ v16bf mul_frag(const float* __restrict__ a,
                                          const float* __restrict__ b,
                                          int k0, int k1) {
  v16bf r;
#pragma unroll
  for (int i = 0; i < 8; ++i) {
    r[i]     = (__bf16)(a[k0 + i] * b[k0 + i]);
    r[i + 8] = (__bf16)(a[k1 + i] * b[k1 + i]);
  }
  return r;
}
__device__ __forceinline__ v16bf cvt_frag(const float* __restrict__ a, int k0, int k1) {
  v16bf r;
#pragma unroll
  for (int i = 0; i < 8; ++i) {
    r[i]     = (__bf16)a[k0 + i];
    r[i + 8] = (__bf16)a[k1 + i];
  }
  return r;
}
// B fragment: lane holds column n, 16 contiguous K values from W^T[n][kb..kb+15]
__device__ __forceinline__ v16bf ldb(const bf16_t* __restrict__ wt, int n, int kb) {
  return *(const v16bf*)(wt + n * 64 + kb);
}
__device__ __forceinline__ v8f wmma_bf16(v16bf a, v16bf b, v8f c) {
  return __builtin_amdgcn_wmma_f32_16x16x32_bf16(false, a, false, b, (short)0, c, false, false);
}

// -------- kernel 0: zero m/denom/accum --------
__global__ void zero_u32(unsigned* __restrict__ p, int n) {
  int i = blockIdx.x * blockDim.x + threadIdx.x;
  if (i < n) p[i] = 0u;
}

// -------- kernel 1: W -> bf16 W^T --------
__global__ void prep_weights(const float* __restrict__ Wq, const float* __restrict__ Wk,
                             const float* __restrict__ Wv, const float* __restrict__ We,
                             bf16_t* __restrict__ WT) {
  int i = blockIdx.x * blockDim.x + threadIdx.x;
  if (i >= 4 * 64 * 64) return;
  int mat = i >> 12;
  int rem = i & 4095;
  int n = rem >> 6, k = rem & 63;
  const float* W = (mat == 0) ? Wq : (mat == 1) ? Wk : (mat == 2) ? Wv : We;
  WT[mat * 4096 + n * 64 + k] = (__bf16)W[k * 64 + n];
}

// -------- kernel 2: q = x @ Wq (one wave per 16-row tile) --------
__global__ void q_gemm(const float* __restrict__ x, const bf16_t* __restrict__ WTq,
                       float* __restrict__ qws) {
  int lane = threadIdx.x;
  int nt = blockIdx.x;
  int row = lane & 15, half = lane >> 4, g0 = half * 16;
  const float* xr = x + (size_t)(nt * 16 + row) * DD;
  v16bf a0 = cvt_frag(xr, g0, g0 + 8);
  v16bf a1 = cvt_frag(xr, 32 + g0, 40 + g0);
#pragma unroll
  for (int jt = 0; jt < 4; ++jt) {
    v8f c = {};
    c = wmma_bf16(a0, ldb(WTq, jt * 16 + row, half * 16), c);
    c = wmma_bf16(a1, ldb(WTq, jt * 16 + row, 32 + half * 16), c);
#pragma unroll
    for (int r = 0; r < 8; ++r)
      qws[(size_t)(nt * 16 + half * 8 + r) * DD + jt * 16 + row] = c[r];
  }
}

// -------- kernel 3: edge pass 1 -> k, edge_out, scores, segment-max --------
__global__ void edge_pass1(const float* __restrict__ x, const long long* __restrict__ eidx,
                           const float* __restrict__ edge_attr, const float* __restrict__ qws,
                           const bf16_t* __restrict__ WTk, const bf16_t* __restrict__ WTe,
                           const float* __restrict__ be, float* __restrict__ scores,
                           unsigned* __restrict__ mkeys, float* __restrict__ edge_out) {
  __shared__ float kl[16][64];
  int lane = threadIdx.x;
  int base = blockIdx.x * 16;
  int row = lane & 15, half = lane >> 4, g0 = half * 16;
  int er = base + row;

  if (base + 16 < EE)  // prefetch next tile's edge_attr rows
    __builtin_prefetch(edge_attr + (size_t)(base + 16) * DD + lane * 32, 0, 0);

  long long nb = eidx[(size_t)EE + er];        // edge_index[1] = nbr
  const float* ea = edge_attr + (size_t)er * DD;
  const float* xn = x + (size_t)nb * DD;
  v16bf a0 = mul_frag(ea, xn, g0, g0 + 8);     // Z tile, K 0..31
  v16bf a1 = mul_frag(ea, xn, 32 + g0, 40 + g0);

#pragma unroll
  for (int jt = 0; jt < 4; ++jt) {
    // k = Z @ Wk  -> LDS for score dot
    v8f ck = {};
    ck = wmma_bf16(a0, ldb(WTk, jt * 16 + row, half * 16), ck);
    ck = wmma_bf16(a1, ldb(WTk, jt * 16 + row, 32 + half * 16), ck);
#pragma unroll
    for (int r = 0; r < 8; ++r) kl[half * 8 + r][jt * 16 + row] = ck[r];
    // edge_out = Z @ We + be -> global
    v8f ce = {};
    ce = wmma_bf16(a0, ldb(WTe, jt * 16 + row, half * 16), ce);
    ce = wmma_bf16(a1, ldb(WTe, jt * 16 + row, 32 + half * 16), ce);
    float bias = be[jt * 16 + row];
#pragma unroll
    for (int r = 0; r < 8; ++r)
      edge_out[(size_t)(base + half * 8 + r) * DD + jt * 16 + row] = ce[r] + bias;
  }
  __syncthreads();

  // scores: each lane handles its edge (row) for 2 heads; q[src] gather hits L2
  long long s = eidx[er];                      // edge_index[0] = src
#pragma unroll
  for (int hh = 0; hh < 2; ++hh) {
    int h = half * 2 + hh;
    const float* qr = qws + (size_t)s * DD + h * 16;
    float acc = 0.f;
#pragma unroll
    for (int c = 0; c < 16; ++c) acc += qr[c] * kl[row][h * 16 + c];
    acc *= 0.25f;                              // 1/sqrt(C=16)
    scores[(size_t)er * 4 + h] = acc;
    atomicMax(&mkeys[(int)s * 4 + h], fkey(acc));
  }
}

// -------- kernel 4: edge pass 2 -> v, exp-weights, scatter-add --------
__global__ void edge_pass2(const float* __restrict__ x, const long long* __restrict__ eidx,
                           const float* __restrict__ edge_attr, const float* __restrict__ scores,
                           const unsigned* __restrict__ mkeys, const bf16_t* __restrict__ WTv,
                           float* __restrict__ denom, float* __restrict__ accum) {
  __shared__ float wl[16][4];
  __shared__ int sl[16];
  int lane = threadIdx.x;
  int base = blockIdx.x * 16;
  int row = lane & 15, half = lane >> 4, g0 = half * 16;
  int er = base + row;

  if (base + 16 < EE)
    __builtin_prefetch(edge_attr + (size_t)(base + 16) * DD + lane * 32, 0, 0);

  long long nb = eidx[(size_t)EE + er];
  const float* ea = edge_attr + (size_t)er * DD;
  const float* xn = x + (size_t)nb * DD;
  v16bf a0 = mul_frag(ea, xn, g0, g0 + 8);
  v16bf a1 = mul_frag(ea, xn, 32 + g0, 40 + g0);

  long long s = eidx[er];
  if (half == 0) sl[row] = (int)s;
#pragma unroll
  for (int hh = 0; hh < 2; ++hh) {
    int h = half * 2 + hh;
    float sc = scores[(size_t)er * 4 + h];
    float m = fdecode(mkeys[(int)s * 4 + h]);
    float w = __expf(sc - m);
    atomicAdd(&denom[(int)s * 4 + h], w);
    wl[row][h] = w;
  }
  __syncthreads();

#pragma unroll
  for (int jt = 0; jt < 4; ++jt) {
    v8f cv = {};
    cv = wmma_bf16(a0, ldb(WTv, jt * 16 + row, half * 16), cv);
    cv = wmma_bf16(a1, ldb(WTv, jt * 16 + row, 32 + half * 16), cv);
#pragma unroll
    for (int r = 0; r < 8; ++r) {
      int e2 = half * 8 + r;
      atomicAdd(&accum[(size_t)sl[e2] * DD + jt * 16 + row], wl[e2][jt] * cv[r]);
    }
  }
}

// -------- kernel 5: out = accum / (denom + 1e-16) --------
__global__ void finalize_nodes(const float* __restrict__ accum, const float* __restrict__ denom,
                               float* __restrict__ out) {
  int i = blockIdx.x * blockDim.x + threadIdx.x;
  if (i >= NN * DD) return;
  int n = i >> 6, col = i & 63, h = col >> 4;
  out[i] = accum[i] / (denom[n * 4 + h] + 1e-16f);
}

extern "C" void kernel_launch(void* const* d_in, const int* in_sizes, int n_in,
                              void* d_out, int out_size, void* d_ws, size_t ws_size,
                              hipStream_t stream) {
  const float*     x    = (const float*)d_in[0];
  const long long* eidx = (const long long*)d_in[1];   // int64 [2,E]: [0]=src, [1]=nbr
  const float*     eatt = (const float*)d_in[2];
  const float*     Wq   = (const float*)d_in[3];
  const float*     Wk   = (const float*)d_in[4];
  const float*     Wv   = (const float*)d_in[5];
  const float*     We   = (const float*)d_in[6];
  const float*     be   = (const float*)d_in[7];

  float* out_nodes = (float*)d_out;                    // [N,64]
  float* out_edges = (float*)d_out + (size_t)NN * DD;  // [E,64]

  char* ws = (char*)d_ws;
  bf16_t*   WT     = (bf16_t*)(ws + OFF_WT);
  float*    qws    = (float*)(ws + OFF_Q);
  float*    scores = (float*)(ws + OFF_SCORES);
  unsigned* mkeys  = (unsigned*)(ws + OFF_MKEYS);
  float*    denom  = (float*)(ws + OFF_DENOM);
  float*    accum  = (float*)(ws + OFF_ACCUM);

  // zero m/denom/accum (contiguous region: (N*4 + N*4 + N*64) u32 words)
  int zwords = NN * 4 + NN * 4 + NN * DD;
  zero_u32<<<(zwords + 255) / 256, 256, 0, stream>>>(mkeys, zwords);
  prep_weights<<<64, 256, 0, stream>>>(Wq, Wk, Wv, We, WT);
  q_gemm<<<NN / 16, 32, 0, stream>>>(x, WT /*WTq at offset 0*/, qws);
  edge_pass1<<<EE / 16, 32, 0, stream>>>(x, eidx, eatt, qws,
                                         WT + 1 * 4096 /*Wk*/, WT + 3 * 4096 /*We*/,
                                         be, scores, mkeys, out_edges);
  edge_pass2<<<EE / 16, 32, 0, stream>>>(x, eidx, eatt, scores, mkeys,
                                         WT + 2 * 4096 /*Wv*/, denom, accum);
  finalize_nodes<<<(NN * DD + 255) / 256, 256, 0, stream>>>(accum, denom, out_nodes);
}
